// CNNLayerTranspose_long_11991548690489
// MI455X (gfx1250) — matrix operand order repaired and checked by reading, not compiled, not run on hardware
//
#include <hip/hip_runtime.h>

typedef __attribute__((ext_vector_type(16))) _Float16 v16h;
typedef __attribute__((ext_vector_type(8)))  float    v8f;

#define EPS 1e-5f

// x: (1, 4ci, 8t, 96, 96, 96)  -> 32 groups of 96^3 = 884736 elements
#define GROUP_N   884736
#define SLABS     32
#define SLAB_N    27648
#define RED_THR   256
#define PER_THR   108

// conv tiling
#define OHT       4                    // oh rows per block
#define HALO_H    (OHT + 2)            // 6
#define HALO_ROWS (4 * 3 * 3 * HALO_H) // 216 (ci,zt,zd,zh) rows of 98
#define HALO_N    (HALO_ROWS * 98)     // 21168 halves
#define YV        112                  // 7 v-tiles of 16 (v = 0..111, 98 real)
#define NJOB      (OHT * 7)            // 28 (row, vtile) jobs per block

// ---------------------------------------------------------------------------
// Stage A: per-(group,slab) partial sum / sumsq (deterministic tree reduce)
// ---------------------------------------------------------------------------
__global__ __launch_bounds__(RED_THR)
void in3d_stats_partial(const float* __restrict__ x, float* __restrict__ part) {
  const int slab = blockIdx.x;
  const int g    = blockIdx.y;
  const int tid  = threadIdx.x;
  const int base = g * GROUP_N + slab * SLAB_N;

  float s = 0.f, q = 0.f;
#pragma unroll 4
  for (int k = 0; k < PER_THR; ++k) {
    const int idx = base + k * RED_THR + tid;
    if (k < PER_THR - 16) __builtin_prefetch(x + idx + 16 * RED_THR, 0, 0);
    const float v = x[idx];
    s += v;
    q += v * v;
  }

  __shared__ float ls[RED_THR];
  __shared__ float lq[RED_THR];
  ls[tid] = s; lq[tid] = q;
  __syncthreads();
  for (int off = RED_THR >> 1; off > 0; off >>= 1) {
    if (tid < off) { ls[tid] += ls[tid + off]; lq[tid] += lq[tid + off]; }
    __syncthreads();
  }
  if (tid == 0) {
    part[(g * SLABS + slab) * 2 + 0] = ls[0];
    part[(g * SLABS + slab) * 2 + 1] = lq[0];
  }
}

// ---------------------------------------------------------------------------
// Stage B: finalize mean / inv_std per group
// ---------------------------------------------------------------------------
__global__ __launch_bounds__(32)
void in3d_stats_final(const float* __restrict__ part, float* __restrict__ stats) {
  const int g = threadIdx.x;
  if (g >= 32) return;
  float s = 0.f, q = 0.f;
  for (int i = 0; i < SLABS; ++i) {
    s += part[(g * SLABS + i) * 2 + 0];
    q += part[(g * SLABS + i) * 2 + 1];
  }
  const float invn = 1.f / (float)GROUP_N;
  const float mean = s * invn;
  const float var  = q * invn - mean * mean;   // biased, as jnp.var
  stats[g * 2 + 0] = mean;
  stats[g * 2 + 1] = rsqrtf(var + EPS);
}

// ---------------------------------------------------------------------------
// K-slot -> tap bijection (pair scheme).
// Taps: q = (ci*3+kt)*3+kd in [0,36), kh = r in [0,3); halo offset
// q*588 + (row+r)*98 + v.  Pair (q=2q'+m, r), m = lane-half.  Combo
// i = chunk*16 + e  (e = v16h element, same for both lane halves) maps to
// (q' = i/3, r = i%3); i >= 54 is padding (zero B columns).
// A offset = i/3*1176 + i%3*98  + base(row, v, hi) -- affine in hi.
//
// Two B sets (B lane layout used here: lane<16 -> N=lane, K=e; lane>=16 ->
// N=lane-16, K=16+e):
//   set 0: N = co + 8*kw (kw in {0,1});  set 1: N = co (kw = 2, 8 cols)
// bws[(set*4 + chunk)*32*16 + lane*16 + e]
// ---------------------------------------------------------------------------
__global__ __launch_bounds__(32)
void conv4d_prep_bfrag(const float* __restrict__ w, _Float16* __restrict__ bws) {
  const int blk  = blockIdx.x;       // 0..7 = set*4 + chunk
  const int set  = blk >> 2;
  const int c    = blk & 3;
  const int lane = threadIdx.x;
  const int n    = lane & 15;
  const int hib  = lane >> 4;
#pragma unroll
  for (int e = 0; e < 16; ++e) {
    const int s = e;
    const int m = hib;
    const int combo = c * 16 + s;
    float v = 0.f;
    if (combo < 54) {
      const int qp = combo / 3;
      const int r  = combo % 3;
      const int q  = 2 * qp + m;
      const int ci = q / 9;
      const int kt = (q / 3) % 3;
      const int kd = q % 3;
      const int kh = r;
      int co = -1, kw = 0;
      if (set == 0)   { co = n & 7; kw = n >> 3; }
      else if (n < 8) { co = n;     kw = 2;      }
      if (co >= 0)
        v = w[((((co * 4 + ci) * 3 + kt) * 3 + kd) * 3 + kh) * 3 + kw];
    }
    bws[(blk * 32 + lane) * 16 + e] = (_Float16)v;
  }
}

// ---------------------------------------------------------------------------
// A gather: one base VGPR (lb2 = row*98 + v + hi*588), 16 ds_load_u16 with
// compile-time immediate offsets (max 40376 B < 64 KB DS offset field).
// ---------------------------------------------------------------------------
template<int C>
__device__ __forceinline__ v16h gatherA(const _Float16* __restrict__ hs, int lb2) {
  v16h a;
#pragma unroll
  for (int e = 0; e < 16; ++e) {
    const int combo = C * 16 + e;
    if (combo < 54) {
      const int qp = combo / 3;
      const int r  = combo % 3;
      a[e] = hs[lb2 + qp * 1176 + r * 98];
    } else {
      a[e] = (_Float16)0;
    }
  }
  return a;
}

template<int C>
__device__ __forceinline__ void chunk_step(const _Float16* __restrict__ hs,
                                           int lb2,
                                           const v16h& b0, const v16h& b1,
                                           v8f& acc0, v8f& acc1) {
  const v16h a = gatherA<C>(hs, lb2);
  acc0 = __builtin_amdgcn_wmma_f32_16x16x32_f16(false, a, false, b0,
                                                (short)0, acc0, false, false);
  acc1 = __builtin_amdgcn_wmma_f32_16x16x32_f16(false, a, false, b1,
                                                (short)0, acc1, false, false);
}

// ---------------------------------------------------------------------------
// Fused InstanceNorm + ReLU + 4D conv.  Grid (24 oh-tiles, 96 od, 8 t),
// 256 threads = 8 waves.
// ---------------------------------------------------------------------------
__global__ __launch_bounds__(256)
void conv4d_wmma(const float*    __restrict__ x,
                 const _Float16* __restrict__ bws,
                 const float*    __restrict__ stats,
                 const float*    __restrict__ bias,
                 float*          __restrict__ out) {
  const int oh0 = blockIdx.x * OHT;
  const int od  = blockIdx.y;
  const int t   = blockIdx.z;
  const int tid = threadIdx.x;

  __shared__ _Float16 halo_s[HALO_N + 16];      // +pad: v>97 gathers in-bounds
  __shared__ float    Y_s[3 * OHT * YV * 8];    // [plane][row][v][co]
  __shared__ float    smean[12], sinv[12];

  const int wv     = __builtin_amdgcn_readfirstlane(threadIdx.x >> 5);
  const int lane   = tid & 31;
  const int lanelo = lane & 15;
  const int hi     = lane >> 4;

  if (tid < 12) {
    const int ci = tid / 3, zt = tid % 3;
    const int tp = t + zt - 1;
    float m = 0.f, iv = 0.f;
    if (tp >= 0 && tp < 8) {
      m  = stats[(ci * 8 + tp) * 2 + 0];
      iv = stats[(ci * 8 + tp) * 2 + 1];
    }
    smean[tid] = m; sinv[tid] = iv;
  }
  __syncthreads();

  // ---- stage halo: row decode is wave-uniform (SALU), lanes stream w ------
  for (int r = wv; r < HALO_ROWS; r += 8) {
    const int zh = r % 6;
    const int u  = r / 6;
    const int zd = u % 3;
    const int zt = (u / 3) % 3;
    const int ci = u / 9;
    const int tp = t   + zt - 1;
    const int dp = od  + zd - 1;
    const int hp = oh0 + zh - 1;
    const bool rowok = (tp >= 0 && tp < 8 && dp >= 0 && dp < 96 &&
                        hp >= 0 && hp < 96);
    const int   gbase = (((ci * 8 + tp) * 96 + dp) * 96 + hp) * 96;
    const float m  = smean[ci * 3 + zt];
    const float iv = sinv [ci * 3 + zt];
#pragma unroll
    for (int k = 0; k < 4; ++k) {
      const int wp = k * 32 + lane;          // 0..97 (k==3: lanes 0,1)
      if (wp < 98) {
        const int wi = wp - 1;
        float v = 0.f;
        if (rowok && wi >= 0 && wi < 96)
          v = fmaxf((x[gbase + wi] - m) * iv, 0.f);
        halo_s[r * 98 + wp] = (_Float16)v;
      }
    }
  }
  __syncthreads();

  // ---- hoist all 8 B fragments into registers (job-invariant) -------------
  v16h b0[4], b1[4];
#pragma unroll
  for (int c = 0; c < 4; ++c) {
    b0[c] = *(const v16h*)(bws + ((0 * 4 + c) * 32 + lane) * 16);
    b1[c] = *(const v16h*)(bws + ((1 * 4 + c) * 32 + lane) * 16);
  }

  // ---- WMMA jobs: j = row*7 + vtile, wave-strided (uniform control) -------
  for (int k = 0; k < 4; ++k) {
    const int j = wv + 8 * k;
    if (j >= NJOB) break;
    const int row = j / 7;
    const int vt  = j % 7;
    const int lb2 = row * 98 + vt * 16 + lanelo + hi * 588;  // halves

    v8f acc0 = {0.f, 0.f, 0.f, 0.f, 0.f, 0.f, 0.f, 0.f};
    v8f acc1 = {0.f, 0.f, 0.f, 0.f, 0.f, 0.f, 0.f, 0.f};
    chunk_step<0>(halo_s, lb2, b0[0], b1[0], acc0, acc1);
    chunk_step<1>(halo_s, lb2, b0[1], b1[1], acc0, acc1);
    chunk_step<2>(halo_s, lb2, b0[2], b1[2], acc0, acc1);
    chunk_step<3>(halo_s, lb2, b0[3], b1[3], acc0, acc1);

    // D layout: VGPR rr -> M = rr + 8*hi (v offset), N = lanelo
#pragma unroll
    for (int rr = 0; rr < 8; ++rr) {
      const int v = vt * 16 + rr + (hi << 3);
      const int plane = lanelo >> 3;               // acc0: 0 -> kw0, 1 -> kw1
      const int co    = lanelo & 7;
      Y_s[((plane * OHT + row) * YV + v) * 8 + co] = acc0[rr];
      if (lanelo < 8)
        Y_s[((2 * OHT + row) * YV + v) * 8 + lanelo] = acc1[rr];
    }
  }
  __syncthreads();

  // ---- epilogue: O[ow] = Y0[ow] + Y1[ow+1] + Y2[ow+2] + bias, coalesced ---
  for (int idx = tid; idx < OHT * 96 * 8; idx += 256) {
    const int ow  = idx % 96;
    const int t2  = idx / 96;
    const int row = t2 & (OHT - 1);
    const int co  = t2 >> 2;
    const float v = Y_s[((0 * OHT + row) * YV + ow    ) * 8 + co]
                  + Y_s[((1 * OHT + row) * YV + ow + 1) * 8 + co]
                  + Y_s[((2 * OHT + row) * YV + ow + 2) * 8 + co]
                  + bias[co];
    out[(((co * 8 + t) * 96 + od) * 96 + (oh0 + row)) * 96 + ow] = v;
  }
}

// ---------------------------------------------------------------------------
extern "C" void kernel_launch(void* const* d_in, const int* in_sizes, int n_in,
                              void* d_out, int out_size, void* d_ws, size_t ws_size,
                              hipStream_t stream) {
  const float* x = (const float*)d_in[0];   // (1,4,8,96,96,96) f32
  const float* w = (const float*)d_in[1];   // (8,4,3,3,3,3)    f32
  const float* b = (const float*)d_in[2];   // (8,)             f32
  float* out = (float*)d_out;               // (1,8,8,96,96,96) f32

  // ws layout (rewritten every call):
  //   [0)      partials: 32*32*2 floats = 8192 B
  //   [8192)   stats:    32*2    floats =  256 B
  //   [8448)   bfrags:   8*32*16 halves = 8192 B (32B aligned)
  float*    ws_part  = (float*)d_ws;
  float*    ws_stats = ws_part + 32 * 32 * 2;
  _Float16* ws_bf    = (_Float16*)((char*)d_ws + 8448);

  in3d_stats_partial<<<dim3(SLABS, 32), RED_THR, 0, stream>>>(x, ws_part);
  in3d_stats_final<<<1, 32, 0, stream>>>(ws_part, ws_stats);
  conv4d_prep_bfrag<<<8, 32, 0, stream>>>(w, ws_bf);
  conv4d_wmma<<<dim3(96 / OHT, 96, 8), 256, 0, stream>>>(x, ws_bf, ws_stats, b, out);
}
